// QKVLoraLayer_59931973648400
// MI455X (gfx1250) — compile-verified, matrix operand order checked by reading, not run on hardware
//
#include <hip/hip_runtime.h>
#include <cstdint>

// ---------------------------------------------------------------------------
// QKV multi-adapter LoRA on MI455X (gfx1250, wave32, WMMA bf16)
//   stage1: H[c]  = bf16( coef ⊙ (X @ Acat[c]) )   M=8192 K=4096 N=512
//   stage2: OUT[c]= H[c] @ Bcat[c]                  M=8192 K=512  N=4096 (f32)
// x converted to bf16 once (64 MiB -> L2-resident across its 12 reuse passes).
// Double-buffered LDS pipeline: A tiles via GLOBAL_LOAD_ASYNC_TO_LDS_B128
// (ASYNCcnt), B tiles transposed through VGPRs, one barrier per K-step.
// ---------------------------------------------------------------------------

typedef __attribute__((ext_vector_type(16))) __bf16 v16bf;
typedef __attribute__((ext_vector_type(8)))  __bf16 v8bf;
typedef __attribute__((ext_vector_type(4)))  __bf16 v4bf;
typedef __attribute__((ext_vector_type(2)))  __bf16 v2bf;
typedef __attribute__((ext_vector_type(8)))  float  v8f;

#define N_ADAPT 8
#define RANK    64
#define BB      4
#define SS      2048
#define DIN     4096
#define DOUT    4096
#define TOKENS  (BB * SS)        /* 8192 */
#define NR      (N_ADAPT * RANK) /* 512  */

// ------------------------------ conversions --------------------------------

__global__ __launch_bounds__(256)
void conv_f32_bf16(const float* __restrict__ src, __bf16* __restrict__ dst, long n) {
  long i = ((long)blockIdx.x * blockDim.x + threadIdx.x) * 4;
  if (i + 3 >= n + 4) return;           // grids are sized exactly; keep uniform
  float4 v = *(const float4*)(src + i);
  v4bf o;
  o[0] = (__bf16)v.x; o[1] = (__bf16)v.y; o[2] = (__bf16)v.z; o[3] = (__bf16)v.w;
  *(v4bf*)(dst + i) = o;
}

// A: [N_ADAPT][DIN][RANK] f32  ->  Acat: [DIN][NR] bf16  (adapters stacked on cols)
__global__ __launch_bounds__(256)
void conv_Acat(const float* __restrict__ A, __bf16* __restrict__ dst) {
  long idx = (long)blockIdx.x * 256 + threadIdx.x;   // over DIN*NR
  int i   = (int)(idx / NR);
  int col = (int)(idx % NR);
  int n   = col >> 6;
  int r   = col & 63;
  dst[idx] = (__bf16)A[((long)n * DIN + i) * RANK + r];
}

// ------------------------------ WMMA GEMM ----------------------------------
// 128x128 block tile, 8 waves in a 2x4 grid, each wave: 64x32 (4x2 fragments).
// K stepped by 32 (one v_wmma_f32_16x16x32_bf16 per fragment per step).
//
// LDS A tile: row-major [128][32] (pad to 40).  A-fragment per ISA 7.12.2:
//   lane = half*16 + m ; VGPR0-3 = K 8*half..+7 ; VGPR4-7 = K 16+8*half..+7
//   -> two 16-byte ds loads per fragment.
// LDS B tile stored transposed [col][k] so B-fragments use the same pattern.

template<int MODE>   // 0: stage1 (scale + bf16 store), 1: stage2 (f32 store)
__global__ __launch_bounds__(256, 2)
void gemm_wmma_bf16(const __bf16* __restrict__ Abase, long aCompStride, int lda,
                    const __bf16* __restrict__ Bbase, long bCompStride, int ldb,
                    int Ksteps,
                    __bf16* __restrict__ Hout, long hCompStride,
                    const float* __restrict__ scaling,
                    const float* __restrict__ masks,
                    float* __restrict__ Cout) {
  __shared__ __bf16 As[2][128][40];   // [buf][row][k]
  __shared__ __bf16 Bs[2][128][40];   // [buf][col][k]  (transposed at staging)

  const int c = blockIdx.z;
  const __bf16* __restrict__ Ag = Abase + (long)c * aCompStride;
  const __bf16* __restrict__ Bg = Bbase + (long)c * bCompStride;

  const int tid  = threadIdx.x;
  const int lane = tid & 31;
  const int wave = tid >> 5;
  const int wr   = wave >> 2;      // 0..1 : 64-row strips
  const int wc   = wave & 3;       // 0..3 : 32-col strips
  const int half = lane >> 4;
  const int l16  = lane & 15;

  const int rowTile = blockIdx.x * 128;
  const int colTile = blockIdx.y * 128;

  // staging coordinates
  const int ar0 = tid >> 2, aks = tid & 3;   // A: row (2 passes of 64), k-seg (8 bf16)
  const int bkp = tid >> 4, bcs = tid & 15;  // B: k-pair 0..15, col-seg (8 cols)

  v8f acc[4][2];
#pragma unroll
  for (int i = 0; i < 4; ++i)
#pragma unroll
    for (int j = 0; j < 2; ++j)
#pragma unroll
      for (int e = 0; e < 8; ++e) acc[i][j][e] = 0.0f;

  v8bf bLo, bHi;   // in-flight B tile (two k rows x 8 cols per thread)

  // ---- A tile: direct global->LDS async DMA (ASYNCcnt), no VGPR bounce ----
  auto stageA_async = [&](int buf, int k0) {
#pragma unroll
    for (int p = 0; p < 2; ++p) {
      int r = ar0 + p * 64;
      const __bf16* g = Ag + (long)(rowTile + r) * lda + k0 + aks * 8;
      // generic LDS pointer: [63:32]=shared aperture, [31:0]=LDS byte address
      unsigned ldsAddr = (unsigned)(unsigned long long)(const void*)&As[buf][r][aks * 8];
      asm volatile("global_load_async_to_lds_b128 %0, %1, off"
                   :: "v"(ldsAddr), "v"((unsigned long long)(const void*)g)
                   : "memory");
    }
  };
  auto loadB = [&](int k0) {                     // issue early: global -> regs
    const __bf16* g = Bg + (long)(k0 + 2 * bkp) * ldb + colTile + bcs * 8;
    bLo = *(const v8bf*)g;
    bHi = *(const v8bf*)(g + ldb);
  };
  auto storeB = [&](int buf) {                   // after compute: regs -> LDS^T
#pragma unroll
    for (int e = 0; e < 8; ++e) {
      v2bf p; p[0] = bLo[e]; p[1] = bHi[e];      // pack (k, k+1) -> one b32
      *(v2bf*)&Bs[buf][bcs * 8 + e][2 * bkp] = p;
    }
  };
  auto compute = [&](int buf) {
    v16bf af[4], bfr[2];
#pragma unroll
    for (int mi = 0; mi < 4; ++mi) {
      int r = wr * 64 + mi * 16 + l16;
      v8bf lo = *(const v8bf*)&As[buf][r][half * 8];
      v8bf hi = *(const v8bf*)&As[buf][r][16 + half * 8];
      af[mi] = __builtin_shufflevector(lo, hi, 0,1,2,3,4,5,6,7,8,9,10,11,12,13,14,15);
    }
#pragma unroll
    for (int ni = 0; ni < 2; ++ni) {
      int cc = wc * 32 + ni * 16 + l16;
      v8bf lo = *(const v8bf*)&Bs[buf][cc][half * 8];
      v8bf hi = *(const v8bf*)&Bs[buf][cc][16 + half * 8];
      bfr[ni] = __builtin_shufflevector(lo, hi, 0,1,2,3,4,5,6,7,8,9,10,11,12,13,14,15);
    }
#pragma unroll
    for (int mi = 0; mi < 4; ++mi)
#pragma unroll
      for (int ni = 0; ni < 2; ++ni)
        acc[mi][ni] = __builtin_amdgcn_wmma_f32_16x16x32_bf16(
            false, af[mi], false, bfr[ni], (short)0, acc[mi][ni], false, false);
  };

  // ---- software pipeline: one barrier per K-step --------------------------
  stageA_async(0, 0);
  loadB(0);
  storeB(0);
  asm volatile("s_wait_asynccnt 0x0" ::: "memory");
  __syncthreads();

  for (int kt = 0; kt < Ksteps; ++kt) {
    const int  cur  = kt & 1;
    const bool more = (kt + 1) < Ksteps;
    if (more) {                       // prefetch tile kt+1 into the other buffer
      stageA_async(cur ^ 1, (kt + 1) * 32);
      loadB((kt + 1) * 32);
    }
    compute(cur);                     // WMMA chain hides the in-flight loads
    if (more) storeB(cur ^ 1);
    asm volatile("s_wait_asynccnt 0x0" ::: "memory");
    __syncthreads();
  }

  // ---- epilogue -----------------------------------------------------------
  // C/D layout: VGPR v -> row = base + half*8 + v ; col = base + (lane&15)
#pragma unroll
  for (int mi = 0; mi < 4; ++mi) {
    const int rbase = rowTile + wr * 64 + mi * 16 + half * 8;
#pragma unroll
    for (int ni = 0; ni < 2; ++ni) {
      const int col = colTile + wc * 32 + ni * 16 + l16;
      if (MODE == 0) {
        // adapter / batch are uniform within one 16x16 fragment
        const int   n    = col >> 6;            // col / RANK
        const int   b    = rbase >> 11;         // row / 2048
        const float coef = scaling[n] * masks[n * BB + b];
        __bf16* __restrict__ Hc = Hout + (long)c * hCompStride;
#pragma unroll
        for (int v = 0; v < 8; ++v)
          Hc[(long)(rbase + v) * NR + col] = (__bf16)(acc[mi][ni][v] * coef);
      } else {
        float* __restrict__ Oc = Cout + (long)c * DOUT;
#pragma unroll
        for (int v = 0; v < 8; ++v)
          Oc[(long)(rbase + v) * (3 * DOUT) + col] = acc[mi][ni][v];
      }
    }
  }
}

// ------------------------------ launcher -----------------------------------

extern "C" void kernel_launch(void* const* d_in, const int* in_sizes, int n_in,
                              void* d_out, int out_size, void* d_ws, size_t ws_size,
                              hipStream_t stream) {
  (void)in_sizes; (void)n_in; (void)out_size; (void)ws_size;

  const float* x       = (const float*)d_in[0];
  // d_in[1] = zero-filled output template (unused)
  const float* masks   = (const float*)d_in[2];   // [N_ADAPT, BB]
  const float* scaling = (const float*)d_in[3];   // [N_ADAPT]
  const float* Amats[3] = { (const float*)d_in[4], (const float*)d_in[6], (const float*)d_in[8] };
  const float* Bmats[3] = { (const float*)d_in[5], (const float*)d_in[7], (const float*)d_in[9] };
  float* out = (float*)d_out;

  // workspace layout (bf16):
  //   xb   : TOKENS*DIN   = 33,554,432
  //   Acat : 3 * DIN*NR   =  6,291,456
  //   Bcat : 3 * NR*DOUT  =  6,291,456
  //   Hs   : 3 * TOKENS*NR= 12,582,912          (~117 MiB total)
  __bf16* xb   = (__bf16*)d_ws;
  __bf16* Acat = xb   + (long)TOKENS * DIN;
  __bf16* Bcat = Acat + 3L * DIN * NR;
  __bf16* Hs   = Bcat + 3L * NR * DOUT;

  const long nx = (long)TOKENS * DIN;            // divisible by 1024
  conv_f32_bf16<<<dim3((unsigned)(nx / 1024)), dim3(256), 0, stream>>>(x, xb, nx);

  for (int cpt = 0; cpt < 3; ++cpt) {
    const long nb = (long)NR * DOUT;             // divisible by 1024
    conv_f32_bf16<<<dim3((unsigned)(nb / 1024)), dim3(256), 0, stream>>>(
        Bmats[cpt], Bcat + cpt * nb, nb);
    conv_Acat<<<dim3((DIN * NR) / 256), dim3(256), 0, stream>>>(
        Amats[cpt], Acat + (long)cpt * DIN * NR);
  }

  // stage 1: H[c] = coef ⊙ (X @ Acat[c]) ; M=8192 N=512 K=4096
  gemm_wmma_bf16<0><<<dim3(TOKENS / 128, NR / 128, 3), dim3(256), 0, stream>>>(
      xb, 0L, DIN,
      Acat, (long)DIN * NR, NR,
      DIN / 32,
      Hs, (long)TOKENS * NR, scaling, masks, nullptr);

  // stage 2: OUT[:, :, c*4096 + :] = H[c] @ Bcat[c] ; M=8192 N=4096 K=512
  gemm_wmma_bf16<1><<<dim3(TOKENS / 128, DOUT / 128, 3), dim3(256), 0, stream>>>(
      Hs, (long)TOKENS * NR, NR,
      Bcat, (long)NR * DOUT, DOUT,
      NR / 32,
      nullptr, 0L, nullptr, nullptr, out);
}
